// GOAL_33998961115437
// MI455X (gfx1250) — matrix operand order repaired and checked by reading, not compiled, not run on hardware
//
#include <hip/hip_runtime.h>
#include <math.h>

typedef __attribute__((ext_vector_type(16))) __bf16 v16bf;
typedef __attribute__((ext_vector_type(8)))  __bf16 v8bf;
typedef __attribute__((ext_vector_type(8)))  float  v8f;

#define EPS_NORM 1e-12f
#define EPS_BN   1e-5f

// ---------------------------------------------------------------- utilities

__global__ void k_fill0(float* __restrict__ p, long long n) {
  long long i = (long long)blockIdx.x * blockDim.x + threadIdx.x;
  long long stride = (long long)gridDim.x * blockDim.x;
  for (; i < n; i += stride) p[i] = 0.0f;
}

// in-degree accumulation for both graphs at once
__global__ void k_deg(const int* __restrict__ dg, const int* __restrict__ dgh,
                      float* __restrict__ deg_g, float* __restrict__ deg_gh, int E) {
  int e = blockIdx.x * blockDim.x + threadIdx.x;
  if (e >= E) return;
  atomicAdd(&deg_g[dg[e]], 1.0f);
  atomicAdd(&deg_gh[dgh[e]], 1.0f);
}

// norm = clip(deg,1)^-0.5  (in place, both graphs)
__global__ void k_norm(float* __restrict__ a, float* __restrict__ b, int N) {
  int i = blockIdx.x * blockDim.x + threadIdx.x;
  if (i >= N) return;
  a[i] = rsqrtf(fmaxf(a[i], 1.0f));
  b[i] = rsqrtf(fmaxf(b[i], 1.0f));
}

// out[row,:] = l2norm( (rowscale ? rowscale[row] : 1) * in[row,:] )
// blockDim.x == D (power of two, <= 256), grid.x == Nrows. In-place safe.
__global__ void k_scale_l2norm(const float* __restrict__ in,
                               const float* __restrict__ rowscale,
                               float* __restrict__ out, int D) {
  __shared__ float red[256];
  int row = blockIdx.x, t = threadIdx.x;
  size_t o = (size_t)row * D + t;
  float v = in[o];
  if (rowscale) v *= rowscale[row];
  red[t] = v * v;
  __syncthreads();
  for (int s = D >> 1; s > 0; s >>= 1) {
    if (t < s) red[t] += red[t + s];
    __syncthreads();
  }
  float inv = 1.0f / fmaxf(sqrtf(red[0]), EPS_NORM);
  out[o] = v * inv;
}

// ------------------------------------------------------------------- SpMM
// agg[dst,:] += norm_src[src] * h[src,:]   (one thread per edge x float4-chunk)
__global__ void k_spmm(const float* __restrict__ h, const int* __restrict__ src,
                       const int* __restrict__ dst, const float* __restrict__ nsrc,
                       float* __restrict__ agg, int E, int D, int cshift) {
  long long idx = (long long)blockIdx.x * blockDim.x + threadIdx.x;
  long long total = (long long)E << cshift;
  if (idx >= total) return;
  int e = (int)(idx >> cshift);
  int c = (int)(idx & ((1 << cshift) - 1));
  int s = src[e], d = dst[e];
  float ns = nsrc[s];
  float4 v = ((const float4*)(h + (size_t)s * D))[c];
  float* o = agg + (size_t)d * D + ((size_t)c << 2);
  atomicAdd(o + 0, v.x * ns);
  atomicAdd(o + 1, v.y * ns);
  atomicAdd(o + 2, v.z * ns);
  atomicAdd(o + 3, v.w * ns);
}

// ------------------------------------------------------------- WMMA GEMM
// C[Nrows,Dout] = (rowscale.*A)[Nrows,K] @ Bw[K,Dout]; fp32 in/out, bf16 MACs.
// Block: 128 threads = 4 waves. Block tile: 64 rows x 64 cols.
// The 64-col W panel is staged once in LDS as bf16 (conflict-free K+8 stride);
// each wave computes a 16x64 strip: 4 WMMAs per 32-K chunk sharing one A frag.
__global__ void __launch_bounds__(128)
k_gemm_bf16(const float* __restrict__ A, const float* __restrict__ Bw,
            const float* __restrict__ rowscale, float* __restrict__ C,
            int Nrows, int K, int Dout) {
  __shared__ __bf16 ldsB[64 * (256 + 8)];          // max K = 256
  const int strideB = K + 8;                        // banks 4*n mod 64: conflict-free
  const int tid    = threadIdx.x;
  const int lane   = tid & 31;
  const int wave   = tid >> 5;
  const int tileM  = (blockIdx.x * 4 + wave) * 16;
  const int tileN0 = blockIdx.y * 64;

  // ---- stage W[:, tileN0 .. tileN0+63] into LDS as bf16 (coalesced) ----
  {
    const int n = tid & 63;
    for (int k = (tid >> 6); k < K; k += 2)
      ldsB[n * strideB + k] = (__bf16)Bw[(size_t)k * Dout + tileN0 + n];
  }
  __syncthreads();

  if (tileM >= Nrows) return;               // wave-uniform: EXEC stays all-1s

  const int sub   = lane & 15;
  const int half  = lane >> 4;              // lane half selects K sub-block
  const int mrow  = tileM + sub;            // A: lane indexes M
  const int aKoff = half * 8;               // A lanes 16-31: K+8
  const int bKoff = half * 8;               // B lanes 16-31: K+8

  const float rs = rowscale ? rowscale[mrow] : 1.0f;
  const float* arow = A + (size_t)mrow * K;

  v8f acc[4];
#pragma unroll
  for (int nt = 0; nt < 4; ++nt) acc[nt] = (v8f){0.f,0.f,0.f,0.f,0.f,0.f,0.f,0.f};

  for (int kc = 0; kc < K; kc += 32) {
    if (kc + 32 < K) __builtin_prefetch(arow + kc + 32, 0, 1);
    // A fragment: elements 0..7 = K kc+aKoff.., elements 8..15 = K kc+16+aKoff..
    const float4 f0 = *(const float4*)(arow + kc + aKoff);
    const float4 f1 = *(const float4*)(arow + kc + aKoff + 4);
    const float4 f2 = *(const float4*)(arow + kc + 16 + aKoff);
    const float4 f3 = *(const float4*)(arow + kc + 16 + aKoff + 4);
    v16bf a;
    a[0]  = (__bf16)(f0.x * rs); a[1]  = (__bf16)(f0.y * rs);
    a[2]  = (__bf16)(f0.z * rs); a[3]  = (__bf16)(f0.w * rs);
    a[4]  = (__bf16)(f1.x * rs); a[5]  = (__bf16)(f1.y * rs);
    a[6]  = (__bf16)(f1.z * rs); a[7]  = (__bf16)(f1.w * rs);
    a[8]  = (__bf16)(f2.x * rs); a[9]  = (__bf16)(f2.y * rs);
    a[10] = (__bf16)(f2.z * rs); a[11] = (__bf16)(f2.w * rs);
    a[12] = (__bf16)(f3.x * rs); a[13] = (__bf16)(f3.y * rs);
    a[14] = (__bf16)(f3.z * rs); a[15] = (__bf16)(f3.w * rs);

#pragma unroll
    for (int nt = 0; nt < 4; ++nt) {
      // B fragment: lane = N column (sub of this 16-wide tile), elems = K slice
      const __bf16* bp = &ldsB[(nt * 16 + sub) * strideB + kc + bKoff];
      v8bf b0 = *(const v8bf*)(bp);
      v8bf b1 = *(const v8bf*)(bp + 16);
      v16bf b;
#pragma unroll
      for (int e = 0; e < 8; ++e) { b[e] = b0[e]; b[8 + e] = b1[e]; }
      acc[nt] = __builtin_amdgcn_wmma_f32_16x16x32_bf16(
          /*neg_a=*/false, a, /*neg_b=*/false, b,
          /*c_mod=*/(short)0, acc[nt], /*reuse_a=*/false, /*reuse_b=*/false);
    }
  }

  // C/D layout: VGPR r -> row tileM+r (lanes 0-15) / tileM+8+r (lanes 16-31)
  const int rbase = tileM + half * 8;
#pragma unroll
  for (int nt = 0; nt < 4; ++nt) {
    const int ncol = tileN0 + nt * 16 + sub;
#pragma unroll
    for (int r = 0; r < 8; ++r)
      C[(size_t)(rbase + r) * Dout + ncol] = acc[nt][r];
  }
}

// ---------------------------------------------------------------- combine
// out = beta*hp + alpha*hi - gamma*l2n(hn) - delta*l2n(norm_gh[row]*t2)
// (the reference's h_n = -(h_n@W) sign is folded into -gamma here)
__global__ void k_combine(const float* __restrict__ hp, const float* __restrict__ hn,
                          const float* __restrict__ hi, const float* __restrict__ t2,
                          const float* __restrict__ norm_gh, float* __restrict__ out,
                          const float* __restrict__ pa, const float* __restrict__ pb,
                          const float* __restrict__ pg, const float* __restrict__ pd,
                          int D) {
  __shared__ float r1[256];
  __shared__ float r2[256];
  int row = blockIdx.x, t = threadIdx.x;
  size_t o = (size_t)row * D + t;
  float vp = hp[o], vn = hn[o], vi = hi[o];
  float v2 = t2[o] * norm_gh[row];
  r1[t] = vn * vn;
  r2[t] = v2 * v2;
  __syncthreads();
  for (int s = D >> 1; s > 0; s >>= 1) {
    if (t < s) { r1[t] += r1[t + s]; r2[t] += r2[t + s]; }
    __syncthreads();
  }
  float in1 = 1.0f / fmaxf(sqrtf(r1[0]), EPS_NORM);
  float in2 = 1.0f / fmaxf(sqrtf(r2[0]), EPS_NORM);
  float a = *pa, b = *pb, g = *pg, d = *pd;
  out[o] = b * vp + a * vi - g * (vn * in1) - d * (v2 * in2);
}

// --------------------------------------------------------------- BatchNorm
__global__ void k_bn_partial(const float* __restrict__ x, float* __restrict__ sums,
                             int Nrows, int D, int rows_per_block) {
  int col = threadIdx.x;
  int r0 = blockIdx.x * rows_per_block;
  int r1 = min(r0 + rows_per_block, Nrows);
  float s = 0.f, s2 = 0.f;
  for (int r = r0; r < r1; ++r) {
    float v = x[(size_t)r * D + col];
    s += v; s2 += v * v;
  }
  atomicAdd(&sums[col], s);
  atomicAdd(&sums[D + col], s2);
}

__global__ void k_bn_apply(float* __restrict__ x, const float* __restrict__ sums,
                           const float* __restrict__ scale, const float* __restrict__ bias,
                           int Nrows, int D) {
  int idx = blockIdx.x * blockDim.x + threadIdx.x;
  if (idx >= Nrows * D) return;
  int col = idx & (D - 1);                  // D is a power of two (256)
  float invN = 1.0f / (float)Nrows;
  float mu = sums[col] * invN;
  float var = sums[D + col] * invN - mu * mu;
  float v = (x[idx] - mu) * rsqrtf(var + EPS_BN) * scale[col] + bias[col];
  x[idx] = fmaxf(v, 0.0f);
}

// ---------------------------------------------------------------- driver

static inline int cdiv(long long a, long long b) { return (int)((a + b - 1) / b); }

extern "C" void kernel_launch(void* const* d_in, const int* in_sizes, int n_in,
                              void* d_out, int out_size, void* d_ws, size_t ws_size,
                              hipStream_t stream) {
  const float* features = (const float*)d_in[0];
  const float* Wmat[3]  = {(const float*)d_in[1], (const float*)d_in[2], (const float*)d_in[3]};
  const float* bnsc[2]  = {(const float*)d_in[4], (const float*)d_in[6]};
  const float* bnbi[2]  = {(const float*)d_in[5], (const float*)d_in[7]};
  const float* alpha = (const float*)d_in[8];
  const float* beta  = (const float*)d_in[9];
  const float* gamma = (const float*)d_in[10];
  const float* delta = (const float*)d_in[11];
  const int* src_g  = (const int*)d_in[12];
  const int* dst_g  = (const int*)d_in[13];
  const int* src_gh = (const int*)d_in[14];
  const int* dst_gh = (const int*)d_in[15];

  const int IN = 128, HID = 256;
  const int N = in_sizes[0] / IN;
  const int E = in_sizes[12];
  int dims[4] = {IN, HID, HID, in_sizes[3] / HID};   // {128,256,256,64}

  // workspace layout (fp32): norm_g[N] | norm_gh[N] | bn sums[512] | 4x N*HID
  float* ws      = (float*)d_ws;
  float* norm_g  = ws;
  float* norm_gh = ws + N;
  float* sums    = ws + 2 * (size_t)N;
  size_t bufElems = (size_t)N * HID;
  float* base = ws + 2 * (size_t)N + 512;
  float* h  = base;
  float* F0 = base + bufElems;
  float* F1 = base + 2 * bufElems;
  float* F2 = base + 3 * bufElems;

  // degrees -> symmetric-norm factors for both graphs
  k_fill0<<<1024, 256, 0, stream>>>(norm_g, 2LL * N);
  k_deg<<<cdiv(E, 256), 256, 0, stream>>>(dst_g, dst_gh, norm_g, norm_gh, E);
  k_norm<<<cdiv(N, 256), 256, 0, stream>>>(norm_g, norm_gh, N);

  // h = l2norm(features)
  k_scale_l2norm<<<N, IN, 0, stream>>>(features, nullptr, h, IN);

  for (int i = 0; i < 3; ++i) {
    int Din = dims[i], Dout = dims[i + 1];
    long long nd_in  = (long long)N * Din;
    long long nd_out = (long long)N * Dout;
    int cs_in  = (Din  == 256) ? 6 : (Din  == 128) ? 5 : 4;
    int cs_out = (Dout == 256) ? 6 : (Dout == 128) ? 5 : 4;

    // agg_p = SpMM_g(norm.*h); agg_n = SpMM_gh(norm.*h)
    k_fill0<<<2048, 256, 0, stream>>>(F0, nd_in);
    k_fill0<<<2048, 256, 0, stream>>>(F1, nd_in);
    long long tot = (long long)E << cs_in;
    k_spmm<<<cdiv(tot, 256), 256, 0, stream>>>(h, src_g,  dst_g,  norm_g,  F0, E, Din, cs_in);
    k_spmm<<<cdiv(tot, 256), 256, 0, stream>>>(h, src_gh, dst_gh, norm_gh, F1, E, Din, cs_in);
    // h_n = l2norm(norm_gh .* agg_n)   (in place)
    k_scale_l2norm<<<N, Din, 0, stream>>>(F1, norm_gh, F1, Din);

    // three WMMA GEMMs; dst-norm of the low-pass branch fused as A row-scale
    dim3 gg(cdiv(cdiv(N, 16), 4), Dout / 64);
    k_gemm_bf16<<<gg, 128, 0, stream>>>(F0, Wmat[i], norm_g, F2, N, Din, Dout);   // hp
    k_gemm_bf16<<<gg, 128, 0, stream>>>(F1, Wmat[i], nullptr, F0, N, Din, Dout);  // hn (F1 free after)
    k_gemm_bf16<<<gg, 128, 0, stream>>>(h,  Wmat[i], nullptr, F1, N, Din, Dout);  // hi (h free after)

    // t2 = SpMM_gh(norm.*hp)  into the freed h buffer
    k_fill0<<<2048, 256, 0, stream>>>(h, nd_out);
    long long tot2 = (long long)E << cs_out;
    k_spmm<<<cdiv(tot2, 256), 256, 0, stream>>>(F2, src_gh, dst_gh, norm_gh, h, E, Dout, cs_out);

    float* dest = (i == 2) ? (float*)d_out : F2;   // in-place over hp is safe (row-local)
    k_combine<<<N, Dout, 0, stream>>>(F2, F0, F1, h, norm_gh, dest,
                                      alpha, beta, gamma, delta, Dout);

    if (i < 2) {
      k_fill0<<<4, 256, 0, stream>>>(sums, 2LL * Dout);
      const int strips = 200;
      int rpb = cdiv(N, strips);
      k_bn_partial<<<strips, Dout, 0, stream>>>(F2, sums, N, Dout, rpb);
      k_bn_apply<<<cdiv(nd_out, 256), 256, 0, stream>>>(F2, sums, bnsc[i], bnbi[i], N, Dout);
      // rotate buffers: new h = F2; recycle old h as next scratch
      float* oldh = h; h = F2; F2 = oldh;
    }
  }
}
